// SelfAttention_13941463842975
// MI455X (gfx1250) — matrix operand order repaired and checked by reading, not compiled, not run on hardware
//
#include <hip/hip_runtime.h>

// ---------------------------------------------------------------------------
// Self-attention (B=8, C=256, H=W=64) for gfx1250 / MI455X.
// Flash-attention structure: never materializes the 537MB energy matrix.
// All GEMMs via v_wmma_f32_16x16x32_bf16 (wave32, 16x16 tiles, K=32).
// S is computed TRANSPOSED (S^T = K^T Q) so the softmax row index lands in
// the WMMA column position -> per-lane softmax state, and the P operand for
// P*V is built with 8 cross-half shuffles instead of an LDS restripe.
// Q is pre-scaled by log2(e) so the softmax runs on raw v_exp_f32 (exp2).
// K/V tiles staged to LDS by the Tensor Data Mover (double-buffered).
// ---------------------------------------------------------------------------

typedef __bf16 v16bf __attribute__((ext_vector_type(16)));
typedef float  v8f   __attribute__((ext_vector_type(8)));
typedef unsigned int u32x4 __attribute__((ext_vector_type(4)));
typedef int    i32x4 __attribute__((ext_vector_type(4)));
typedef int    i32x8 __attribute__((ext_vector_type(8)));
typedef float  f32x4 __attribute__((ext_vector_type(4)));

#define DEVI __device__ __forceinline__

#if defined(__HIP_DEVICE_COMPILE__) && __has_builtin(__builtin_amdgcn_tensor_load_to_lds) && __has_builtin(__builtin_amdgcn_s_wait_tensorcnt)
#define USE_TDM 1
#else
#define USE_TDM 0
#endif

constexpr int Bn  = 8;      // batch
constexpr int Cc  = 256;    // channels
constexpr int Nn  = 4096;   // H*W
constexpr int Cqk = 32;     // query/key channels
constexpr float LOG2E = 1.44269504088896340736f;

// One WMMA operand: 16 bf16 per lane (32 bytes) = A(16x32) or B(32x16) tile.
union ABOp { v16bf bf; u32x4 q[2]; unsigned short u[16]; };

DEVI unsigned short f32_bf16(float f) {
  unsigned int x = __float_as_uint(f);
  x += 0x7FFFu + ((x >> 16) & 1u);          // round-to-nearest-even
  return (unsigned short)(x >> 16);
}

// Pack two f32 into one dword of bf16 (low = a, high = b): one v_perm_b32.
DEVI unsigned pack2_bf16(float a, float b) {
#if defined(__HIP_DEVICE_COMPILE__) && __has_builtin(__builtin_amdgcn_perm)
  return __builtin_amdgcn_perm(__float_as_uint(b) + 0x8000u,
                               __float_as_uint(a) + 0x8000u, 0x07060302u);
#else
  return (unsigned)f32_bf16(a) | ((unsigned)f32_bf16(b) << 16);
#endif
}

DEVI float fast_exp2(float x) {
#if defined(__HIP_DEVICE_COMPILE__) && __has_builtin(__builtin_amdgcn_exp2f)
  return __builtin_amdgcn_exp2f(x);         // raw v_exp_f32
#else
  return exp2f(x);
#endif
}

DEVI v8f v8f_zero() {
  v8f z;
#pragma unroll
  for (int i = 0; i < 8; ++i) z[i] = 0.0f;
  return z;
}

DEVI v8f wmma_bf16(const ABOp& a, const ABOp& b, v8f c) {
  // D(16x16 f32) = A(16x32 bf16) x B(32x16 bf16) + C
  return __builtin_amdgcn_wmma_f32_16x16x32_bf16(false, a.bf, false, b.bf,
                                                 (short)0, c, false, false);
}

DEVI u32x4 pack8_bf16(const float* v) {
  u32x4 p;
#pragma unroll
  for (int k = 0; k < 4; ++k) p[k] = pack2_bf16(v[2 * k], v[2 * k + 1]);
  return p;
}

#if USE_TDM
// ---- Tensor Data Mover descriptor builders (CDNA5 ISA) --------
DEVI u32x4 tdm_g0(unsigned lds_addr, const void* gaddr) {
  const unsigned long long ga = (unsigned long long)(size_t)gaddr;
  u32x4 g;
  g[0] = 1u;                                          // count=1, user mode
  g[1] = lds_addr;                                    // lds_addr[31:0]
  g[2] = (unsigned)ga;                                // global_addr[31:0]
  g[3] = (unsigned)((ga >> 32) & 0x1FFFFFFu)          // global_addr[56:32]
       | (2u << 30);                                  // type=2 ("image")
  return g;
}

// 2D tile: tile_dim0 (contiguous, elements) x tile_dim1 rows, row stride in
// elements; tensor dims set to the tile (tiles are always fully in range).
DEVI i32x8 tdm_g1_2d(unsigned tile_d0, unsigned tile_d1, unsigned stride0) {
  const unsigned td0 = tile_d0, td1 = tile_d1;        // tensor dims == tile
  i32x8 g;
  g[0] = (int)(1u << 16);                             // data_size=1 -> 2 bytes
  g[1] = (int)((td0 & 0xFFFFu) << 16);                // tensor_dim0[15:0]
  g[2] = (int)((td0 >> 16) | ((td1 & 0xFFFFu) << 16));// dim0[31:16] dim1[15:0]
  g[3] = (int)((td1 >> 16) | (tile_d0 << 16));        // dim1[31:16] tile_dim0
  g[4] = (int)(tile_d1 & 0xFFFFu);                    // tile_dim1 (tile_dim2=0)
  g[5] = (int)stride0;                                // tensor_dim0_stride lo
  g[6] = 0;                                           // stride0 hi / dim1_stride lo
  g[7] = 0;                                           // dim1_stride hi
  return g;
}

DEVI void tdm_load(unsigned lds_addr, const void* gaddr, i32x8 g1) {
  const i32x4 z4 = (i32x4)0;
  const i32x8 z8 = (i32x8)0;
  // 6-arg form (clang-23 / therock headers): groups 0..3 + extra + cpol
  __builtin_amdgcn_tensor_load_to_lds(tdm_g0(lds_addr, gaddr), g1, z4, z4, z8, 0);
}
#endif

// ---------------------------------------------------------------------------
// Kernel 1: x[B][C][N] f32  ->  xT[B][N][C] bf16   (tiled 32x32 transpose)
// ---------------------------------------------------------------------------
__global__ __launch_bounds__(256) void pack_x_kernel(
    const float* __restrict__ x, unsigned short* __restrict__ xT) {
  __shared__ __align__(16) unsigned short tile[32][33];
  const int b  = blockIdx.z;
  const int c0 = blockIdx.y * 32;
  const int n0 = blockIdx.x * 32;
  const int tx = threadIdx.x;   // 0..31
  const int ty = threadIdx.y;   // 0..7
  const float* xb = x + ((size_t)b * Cc + c0) * Nn + n0;
#pragma unroll
  for (int i = 0; i < 4; ++i) {
    const int row = ty + i * 8;                       // c within tile
    tile[row][tx] = f32_bf16(xb[(size_t)row * Nn + tx]);
  }
  __syncthreads();
  unsigned short* xTb = xT + ((size_t)b * Nn + n0) * Cc + c0;
#pragma unroll
  for (int i = 0; i < 4; ++i) {
    const int row = ty + i * 8;                       // n within tile
    xTb[(size_t)row * Cc + tx] = tile[tx][row];
  }
}

// ---------------------------------------------------------------------------
// Kernel 2: f32 -> bf16 flat convert (weights)
// ---------------------------------------------------------------------------
__global__ __launch_bounds__(256) void cvt_kernel(
    const float* __restrict__ src, unsigned short* __restrict__ dst, int n) {
  const int i = blockIdx.x * 256 + threadIdx.x;
  if (i < n) dst[i] = f32_bf16(src[i]);
}

// ---------------------------------------------------------------------------
// Kernel 3: generic 1x1-conv projection GEMM.
//   D(n-tile 16 x o-tile 16) = ( sum_c xT[b][n][c]*W[o][c] + bias[o] ) * oscale
// OMAJOR=false : out[b][n][o] bf16   (for qT / kT, [B][N][32])
// OMAJOR=true  : out[b][o][n] bf16/f32 (for V [B][C][N] and final output)
// One wave per 16x16 output tile, 8 waves per block.
// ---------------------------------------------------------------------------
template <typename OutT, bool OMAJOR>
__global__ __launch_bounds__(256) void proj_kernel(
    const unsigned short* __restrict__ xT, int Cin,
    const unsigned short* __restrict__ Wb,
    const float* __restrict__ bias,
    OutT* __restrict__ out, int O, float oscale) {
  const int lane = threadIdx.x & 31;
  const int wid  = blockIdx.x * 8 + (threadIdx.x >> 5);
  const int ntiles = Nn >> 4;
  const int otiles = O >> 4;
  const int nt = wid % ntiles;
  const int ot = (wid / ntiles) % otiles;
  const int b  = wid / (ntiles * otiles);
  const int half = lane >> 4, li = lane & 15;
  const int n0 = nt * 16, o0 = ot * 16;

  const unsigned short* arow = xT + (size_t)(b * Nn + n0 + li) * Cin + half * 8;
  const unsigned short* brow = Wb + (size_t)(o0 + li) * Cin + half * 8;

  v8f acc = v8f_zero();
  for (int cb = 0; cb < Cin; cb += 32) {
    ABOp a, w;
    a.q[0] = *(const u32x4*)(arow + cb);
    a.q[1] = *(const u32x4*)(arow + cb + 16);
    w.q[0] = *(const u32x4*)(brow + cb);
    w.q[1] = *(const u32x4*)(brow + cb + 16);
    acc = wmma_bf16(a, w, acc);
  }

  const float bs = bias[o0 + li];
  float res[8];
#pragma unroll
  for (int r = 0; r < 8; ++r) res[r] = (acc[r] + bs) * oscale;

  if constexpr (OMAJOR) {
    const size_t base = (size_t)(b * O + o0 + li) * Nn + n0 + half * 8;
    if constexpr (sizeof(OutT) == 2) {
      *(u32x4*)((unsigned short*)out + base) = pack8_bf16(res);
    } else {
      f32x4 lo, hi;
#pragma unroll
      for (int k = 0; k < 4; ++k) { lo[k] = res[k]; hi[k] = res[k + 4]; }
      *(f32x4*)((float*)out + base)     = lo;
      *(f32x4*)((float*)out + base + 4) = hi;
    }
  } else {
    unsigned short* po = (unsigned short*)out;
#pragma unroll
    for (int r = 0; r < 8; ++r)
      po[(size_t)(b * Nn + n0 + half * 8 + r) * O + o0 + li] = f32_bf16(res[r]);
  }
}

// ---------------------------------------------------------------------------
// Kernel 4: flash attention with transposed-S trick (softmax in exp2 domain;
// q was pre-scaled by log2(e) in its projection).
// ---------------------------------------------------------------------------
__global__ __launch_bounds__(256) void flash_kernel(
    const unsigned short* __restrict__ qT,   // [B][N][32] (log2e-scaled)
    const unsigned short* __restrict__ kT,   // [B][N][32]
    const unsigned short* __restrict__ vv,   // [B][C][N]
    unsigned short* __restrict__ aoT)        // [B][N][C]
{
#if USE_TDM
  __shared__ __align__(16) unsigned short lds_k[2][32][32];   // [buf][j][c]
  __shared__ __align__(16) unsigned short lds_v[2][256][32];  // [buf][c][j]
#else
  __shared__ __align__(16) unsigned short lds_k[1][32][32];
  __shared__ __align__(16) unsigned short lds_v[1][256][32];
#endif

  const int tid  = threadIdx.x;
  const int wv   = tid >> 5;
  const int lane = tid & 31;
  const int half = lane >> 4, li = lane & 15;
  const int blk  = blockIdx.x;
  const int b    = blk >> 5;                       // 32 blocks per batch
  const int i0   = ((blk & 31) * 8 + wv) * 16;     // query tile origin

  // B operand for S^T = K^T Q: column i = i0+li, rows c = 8*half+s, 16+8*half+(s-8) (fixed).
  ABOp bq_op;
  {
    const unsigned short* qrow = qT + (size_t)(b * Nn + i0 + li) * Cqk + half * 8;
    bq_op.q[0] = *(const u32x4*)(qrow);
    bq_op.q[1] = *(const u32x4*)(qrow + 16);
  }

  v8f acc[16];                                     // PV accumulators: 16 c-tiles
#pragma unroll
  for (int ct = 0; ct < 16; ++ct) acc[ct] = v8f_zero();

  float m_i = -1e30f, l_i = 0.0f;                  // softmax state for col i=li

  const unsigned short* kTb = kT + (size_t)b * Nn * Cqk;
  const unsigned short* vb  = vv + (size_t)b * Cc * Nn;
  const int steps = Nn / 32;

#if USE_TDM
  const i32x8 g1k = tdm_g1_2d(32u, 32u, (unsigned)Cqk);   // 32x32, stride 32
  const i32x8 g1v = tdm_g1_2d(32u, 256u, (unsigned)Nn);   // 32x256, stride 4096
  if (wv == 0) {
    tdm_load((unsigned)(size_t)&lds_k[0][0][0], kTb, g1k);
    tdm_load((unsigned)(size_t)&lds_v[0][0][0], vb,  g1v);
  }
#endif

  for (int n = 0; n < steps; ++n) {
    const int j0 = n * 32;
#if USE_TDM
    const int bsel = n & 1;
    if (wv == 0) {
      if (n + 1 < steps) {   // issue next tile's DMA, then wait current done
        tdm_load((unsigned)(size_t)&lds_k[bsel ^ 1][0][0],
                 kTb + (size_t)(j0 + 32) * Cqk, g1k);
        tdm_load((unsigned)(size_t)&lds_v[bsel ^ 1][0][0],
                 vb + (j0 + 32), g1v);
        __builtin_amdgcn_s_wait_tensorcnt(2);
      } else {
        __builtin_amdgcn_s_wait_tensorcnt(0);
      }
    }
    __syncthreads();          // current buffer ready for all waves
#else
    const int bsel = 0;
    __syncthreads();          // previous iteration's consumers are done
    if (tid < 128) {          // k tile: 32 rows x 32 bf16
      const int row = tid >> 2, ch = (tid & 3) * 8;
      *(u32x4*)&lds_k[0][row][ch] =
          *(const u32x4*)(kTb + (size_t)(j0 + row) * Cqk + ch);
    }
#pragma unroll
    for (int it = 0; it < 4; ++it) {               // v tile: 256 rows x 32 bf16
      const int idx = tid + it * 256;
      const int row = idx >> 2, ch = (idx & 3) * 8;
      *(u32x4*)&lds_v[0][row][ch] =
          *(const u32x4*)(vb + (size_t)row * Nn + j0 + ch);
    }
    if (j0 + 32 < Nn) {
      __builtin_prefetch(vb + (size_t)tid * Nn + j0 + 32, 0, 0);
      if (tid < 32)
        __builtin_prefetch(kTb + (size_t)(j0 + 32 + tid) * Cqk, 0, 0);
    }
    __syncthreads();
#endif

    // ---- S^T = K^T Q : rows j (A = k tile from LDS), cols i (B = q) ----
    ABOp ak0, ak1;
    ak0.q[0] = *(const u32x4*)&lds_k[bsel][li][half * 8];
    ak0.q[1] = *(const u32x4*)&lds_k[bsel][li][16 + half * 8];
    ak1.q[0] = *(const u32x4*)&lds_k[bsel][li + 16][half * 8];
    ak1.q[1] = *(const u32x4*)&lds_k[bsel][li + 16][16 + half * 8];
    v8f s0 = wmma_bf16(ak0, bq_op, v8f_zero());    // rows j = 8*half + r
    v8f s1 = wmma_bf16(ak1, bq_op, v8f_zero());    // rows j = 16 + 8*half + r

    // ---- per-lane online softmax (exp2 domain) for column i = i0+li ----
    float mx = fmaxf(s0[0], s1[0]);
#pragma unroll
    for (int r = 1; r < 8; ++r) mx = fmaxf(mx, fmaxf(s0[r], s1[r]));
    mx = fmaxf(mx, __shfl_xor(mx, 16));            // both halves of column i
    const float mnew = fmaxf(m_i, mx);
    const float corr = fast_exp2(m_i - mnew);
    m_i = mnew;

    unsigned pk0[4], pk1[4];
    float rs = 0.0f;
#pragma unroll
    for (int rr = 0; rr < 4; ++rr) {
      const float p0a = fast_exp2(s0[2 * rr]     - mnew);
      const float p0b = fast_exp2(s0[2 * rr + 1] - mnew);
      const float p1a = fast_exp2(s1[2 * rr]     - mnew);
      const float p1b = fast_exp2(s1[2 * rr + 1] - mnew);
      rs += (p0a + p0b) + (p1a + p1b);
      pk0[rr] = pack2_bf16(p0a, p0b);              // v_perm_b32 pack
      pk1[rr] = pack2_bf16(p1a, p1b);
    }
    rs += __shfl_xor(rs, 16);
    l_i = l_i * corr + rs;

    // ---- rescale accumulators (per-lane scalar factor) ----
#pragma unroll
    for (int ct = 0; ct < 16; ++ct)
#pragma unroll
      for (int r = 0; r < 8; ++r) acc[ct][r] *= corr;

    // ---- assemble P as B operand: col i = li, rows j = half*16 + e ----
    ABOp bp;
    {
      u32x4 t0, t1, x0, x1;
#pragma unroll
      for (int k = 0; k < 4; ++k) {
        t0[k] = pk0[k];
        t1[k] = pk1[k];
        x0[k] = (unsigned)__shfl_xor((int)pk0[k], 16);
        x1[k] = (unsigned)__shfl_xor((int)pk1[k], 16);
      }
      bp.q[0] = half ? x1 : t0;    // j = half*16 + 0..7
      bp.q[1] = half ? t1 : x0;    // j = half*16 + 8..15
    }

    // ---- PV: 16 c-tiles, A = v[c][j] rows from LDS ----
#pragma unroll
    for (int ct = 0; ct < 16; ++ct) {
      ABOp av;
      av.q[0] = *(const u32x4*)&lds_v[bsel][ct * 16 + li][half * 16];
      av.q[1] = *(const u32x4*)&lds_v[bsel][ct * 16 + li][half * 16 + 8];
      acc[ct] = wmma_bf16(av, bp, acc[ct]);
    }

#if USE_TDM
    __syncthreads();          // consumers done before buffer is re-filled
#endif
  }

  // ---- epilogue: divide by row sum, store attnOutT[b][i][c] (b128) ----
  const float inv = 1.0f / l_i;
  unsigned short* ao = aoT + (size_t)(b * Nn + i0 + li) * Cc;
#pragma unroll
  for (int ct = 0; ct < 16; ++ct) {
    float res[8];
#pragma unroll
    for (int r = 0; r < 8; ++r) res[r] = acc[ct][r] * inv;  // c = ct*16+8*half+r
    *(u32x4*)(ao + ct * 16 + 8 * half) = pack8_bf16(res);
  }
}

// ---------------------------------------------------------------------------
// Host launcher
// ---------------------------------------------------------------------------
extern "C" void kernel_launch(void* const* d_in, const int* in_sizes, int n_in,
                              void* d_out, int out_size, void* d_ws, size_t ws_size,
                              hipStream_t stream) {
  (void)in_sizes; (void)n_in; (void)out_size; (void)ws_size;
  const float* x  = (const float*)d_in[0];
  const float* Wq = (const float*)d_in[1];
  const float* bq = (const float*)d_in[2];
  const float* Wk = (const float*)d_in[3];
  const float* bk = (const float*)d_in[4];
  const float* Wv = (const float*)d_in[5];
  const float* bv = (const float*)d_in[6];
  const float* Wf = (const float*)d_in[7];
  const float* bf = (const float*)d_in[8];

  char* ws = (char*)d_ws;                    // ~54.9 MB total
  unsigned short* xT  = (unsigned short*)(ws);              // [B][N][C]   16 MiB
  unsigned short* qT  = (unsigned short*)(ws + 16777216);   // [B][N][32]   2 MiB
  unsigned short* kT  = (unsigned short*)(ws + 18874368);   // [B][N][32]   2 MiB
  unsigned short* vv  = (unsigned short*)(ws + 20971520);   // [B][C][N]   16 MiB
  unsigned short* aoT = (unsigned short*)(ws + 37748736);   // [B][N][C]   16 MiB
  unsigned short* wqb = (unsigned short*)(ws + 54525952);
  unsigned short* wkb = (unsigned short*)(ws + 54542336);
  unsigned short* wvb = (unsigned short*)(ws + 54558720);
  unsigned short* wfb = (unsigned short*)(ws + 54689792);

  // 1) pack x -> xT bf16, weights -> bf16
  pack_x_kernel<<<dim3(Nn / 32, Cc / 32, Bn), dim3(32, 8), 0, stream>>>(x, xT);
  cvt_kernel<<<(Cqk * Cc + 255) / 256, 256, 0, stream>>>(Wq, wqb, Cqk * Cc);
  cvt_kernel<<<(Cqk * Cc + 255) / 256, 256, 0, stream>>>(Wk, wkb, Cqk * Cc);
  cvt_kernel<<<(Cc * Cc + 255) / 256, 256, 0, stream>>>(Wv, wvb, Cc * Cc);
  cvt_kernel<<<(Cc * Cc + 255) / 256, 256, 0, stream>>>(Wf, wfb, Cc * Cc);

  // 2) projections (one 16x16 tile per wave, 8 waves per block)
  const int waves_qk = Bn * (Cqk / 16) * (Nn / 16);   // 4096
  const int waves_cc = Bn * (Cc / 16) * (Nn / 16);    // 32768
  proj_kernel<unsigned short, false>
      <<<waves_qk / 8, 256, 0, stream>>>(xT, Cc, wqb, bq, qT, Cqk, LOG2E);
  proj_kernel<unsigned short, false>
      <<<waves_qk / 8, 256, 0, stream>>>(xT, Cc, wkb, bk, kT, Cqk, 1.0f);
  proj_kernel<unsigned short, true>
      <<<waves_cc / 8, 256, 0, stream>>>(xT, Cc, wvb, bv, vv, Cc, 1.0f);

  // 3) flash attention (8 query tiles per block)
  flash_kernel<<<Bn * (Nn / 16) / 8, 256, 0, stream>>>(qT, kT, vv, aoT);

  // 4) final projection -> f32 output [B][C][H][W]
  proj_kernel<float, true>
      <<<waves_cc / 8, 256, 0, stream>>>(aoT, Cc, wfb, bf, (float*)d_out, Cc, 1.0f);
}
